// GumbelQuantize_21620865368349
// MI455X (gfx1250) — matrix operand order repaired and checked by reading, not compiled, not run on hardware
//
#include <hip/hip_runtime.h>
#include <hip/hip_bf16.h>
#include <math.h>

typedef __attribute__((ext_vector_type(16))) __bf16 v16bf;
typedef __attribute__((ext_vector_type(8)))  __bf16 bf16x8;
typedef __attribute__((ext_vector_type(4)))  __bf16 bf16x4;
typedef __attribute__((ext_vector_type(8)))  float  v8f;

#define NBATCH 8
#define CC     256
#define HW     4096
#define NEMB   4096
#define DEMB   64
#define KLW    0.0005f

#define BROW   264                    // padded LDS row stride (bf16 elems) = 528B
#define NTILES (NEMB / 16)            // 256
#define MTILES ((NBATCH * HW) / 16)   // 2048
#define ZQ_ELEMS ((size_t)NBATCH * DEMB * HW)  // 2097152

__global__ __launch_bounds__(256)
void gumbelq_fused(const float* __restrict__ z,
                   const float* __restrict__ pw,
                   const float* __restrict__ pb,
                   const float* __restrict__ ew,
                   const float* __restrict__ uu,
                   float* __restrict__ out,
                   float* __restrict__ klpart)
{
  __shared__ __bf16 Bs[2][16 * BROW];   // double-buffered proj_w tile, bf16
  __shared__ float  Us[8][16 * 17];     // per-wave u-tile transpose buffer

  const int t  = threadIdx.x;
  const int wv = t >> 5;
  const int l  = t & 31;
  const int lo = l & 15;
  const int hi = l >> 4;

  const int mt  = blockIdx.x * 8 + wv;  // M-tile id (0..2047)
  const int p0  = mt * 16;
  const int b   = p0 >> 12;             // 4096 pixels per image
  const int hw0 = p0 & 4095;

  // ---- A tile: 16 pixels x 256 channels, f32 -> bf16, WMMA 16-bit A layout ----
  v16bf A[8];
  {
    const int kb = hi * 8;
    const float* zb = z + (size_t)b * CC * HW + hw0 + lo;
#pragma unroll
    for (int kc = 0; kc < 8; ++kc) {
      v16bf a;
#pragma unroll
      for (int j = 0; j < 8; ++j) {
        a[j]     = (__bf16)zb[(size_t)(kc * 32 + kb + j) * HW];
        a[8 + j] = (__bf16)zb[(size_t)(kc * 32 + 16 + kb + j) * HW];
      }
      A[kc] = a;
    }
  }

  // per-lane online-softmax / argmax state for 8 pixel rows
  float m_[8], S_[8], T_[8], gm_[8];
  int   gi_[8];
#pragma unroll
  for (int r = 0; r < 8; ++r) {
    m_[r] = -__builtin_inff(); S_[r] = 0.f; T_[r] = 0.f;
    gm_[r] = -__builtin_inff(); gi_[r] = 0;
  }

  // running pointers: all hot-loop loads become base + immediate offset
  const int row = t >> 4, seg = t & 15;
  const float4* pwp = (const float4*)(pw + (size_t)(16 + row) * CC + seg * 16);
  const float*  up  = uu + ((size_t)(b * NEMB + hi)) * HW + hw0 + lo;  // code stride folded in
  const float*  pbp = pb + lo;
  __bf16* const bdst0 = &Bs[0][row * BROW + seg * 16];
  __bf16* const bdst1 = &Bs[1][row * BROW + seg * 16];

  // stage first proj_w tile (f32 -> bf16) cooperatively
  {
    const float4* src = (const float4*)(pw + (size_t)row * CC + seg * 16);
#pragma unroll
    for (int q = 0; q < 4; ++q) {
      float4 v = src[q];
      bf16x4 o = { (__bf16)v.x, (__bf16)v.y, (__bf16)v.z, (__bf16)v.w };
      *(bf16x4*)(bdst0 + q * 4) = o;
    }
  }
  __syncthreads();

  for (int nt = 0; nt < NTILES; ++nt) {
    const int n0  = nt * 16;
    const int cur = nt & 1;

    // stage next B tile into the other buffer (overlaps with compute below)
    if (nt + 1 < NTILES) {
      __bf16* d = cur ? bdst0 : bdst1;
#pragma unroll
      for (int q = 0; q < 4; ++q) {
        float4 v = pwp[q];
        bf16x4 o = { (__bf16)v.x, (__bf16)v.y, (__bf16)v.z, (__bf16)v.w };
        *(bf16x4*)(d + q * 4) = o;
      }
    }
    pwp += 16 * CC / 4;   // advance one N-tile

    // coalesced u loads (pixel-contiguous, immediate offsets) -> LDS transpose
#pragma unroll
    for (int j = 0; j < 8; ++j) {
      const float val = up[(size_t)(2 * j) * HW];   // code-in-tile = 2j + hi
      Us[wv][lo * 17 + 2 * j + hi] = val;           // [pixel][code], padded rows
    }
    up += (size_t)16 * HW;

    // 16x16 logits tile: K=256 as 8 x (16x16x32) bf16 WMMA
    v8f acc = {0.f, 0.f, 0.f, 0.f, 0.f, 0.f, 0.f, 0.f};
#pragma unroll
    for (int kc = 0; kc < 8; ++kc) {
      const __bf16* bp = &Bs[cur][lo * BROW + kc * 32 + (hi << 3)];
      bf16x8 b0 = *(const bf16x8*)bp;
      bf16x8 b1 = *(const bf16x8*)(bp + 16);
      v16bf bb = __builtin_shufflevector(b0, b1,
                   0,1,2,3,4,5,6,7,8,9,10,11,12,13,14,15);
      acc = __builtin_amdgcn_wmma_f32_16x16x32_bf16(
              false, A[kc], false, bb, (short)0, acc, false, false);
    }

    // bias + gumbel + online softmax / argmax update
    const int   n  = n0 + lo;
    const float bn = *pbp;
    pbp += 16;
#pragma unroll
    for (int r = 0; r < 8; ++r) {
      const float logit = acc[r] + bn;                       // pixel r+8*hi, code n
      const float uval  = Us[wv][(r + 8 * hi) * 17 + lo];
      const float g = -__logf(-__logf(uval + 1e-20f) + 1e-20f);
      const float y = logit + g;                             // TEMP == 1
      if (y > gm_[r]) { gm_[r] = y; gi_[r] = n; }
      const float nm = fmaxf(m_[r], logit);
      const float e1 = __expf(m_[r] - nm);
      const float e2 = __expf(logit - nm);
      S_[r] = S_[r] * e1 + e2;
      T_[r] = T_[r] * e1 + e2 * logit;
      m_[r] = nm;
    }
    __syncthreads();
  }

  // merge online-softmax/argmax state across the 16 n-partitions (16-lane groups)
#pragma unroll
  for (int r = 0; r < 8; ++r) {
#pragma unroll
    for (int off = 1; off < 16; off <<= 1) {
      const float m2  = __shfl_xor(m_[r],  off, 32);
      const float S2  = __shfl_xor(S_[r],  off, 32);
      const float T2  = __shfl_xor(T_[r],  off, 32);
      const float gm2 = __shfl_xor(gm_[r], off, 32);
      const int   gi2 = __shfl_xor(gi_[r], off, 32);
      const float nm = fmaxf(m_[r], m2);
      const float e1 = __expf(m_[r] - nm);
      const float e2 = __expf(m2   - nm);
      S_[r] = S_[r] * e1 + S2 * e2;
      T_[r] = T_[r] * e1 + T2 * e2;
      m_[r] = nm;
      if (gm2 > gm_[r] || (gm2 == gm_[r] && gi2 < gi_[r])) { gm_[r] = gm2; gi_[r] = gi2; }
    }
  }

  // KL partial (per M-tile) + argmax indices
  float klsum = 0.f;
#pragma unroll
  for (int r = 0; r < 8; ++r)
    klsum += T_[r] / S_[r] - m_[r] - __logf(S_[r]) + 8.31776617f;  // + log(4096)
  const float kltot = klsum + __shfl_xor(klsum, 16, 32);
  if (l == 0) klpart[mt] = kltot;

  if (lo == 0) {
#pragma unroll
    for (int r = 0; r < 8; ++r)
      out[ZQ_ELEMS + 1 + p0 + r + 8 * hi] = (float)gi_[r];
  }

  // z_q = embed_w[argmax] ; lanes -> pixels so stores are 64B-contiguous
  int sel = gi_[0];
#pragma unroll
  for (int r = 1; r < 8; ++r) if ((l & 7) == r) sel = gi_[r];
  const int src = (l & 7) | (((l >> 3) & 1) << 4);
  const int gip = __shfl(sel, src, 32);        // code index for pixel (l&15)

  const float* ewp  = ew + (size_t)gip * DEMB;
  float*       outp = out + (size_t)b * DEMB * HW + hw0 + lo;
#pragma unroll
  for (int it = 0; it < 32; ++it) {
    const int d = (it << 1) | hi;
    outp[(size_t)d * HW] = ewp[d];
  }
}

__global__ __launch_bounds__(256)
void kl_reduce(const float* __restrict__ klpart, float* __restrict__ out)
{
  __shared__ float sm[256];
  float s = 0.f;
  for (int i = threadIdx.x; i < MTILES; i += 256) s += klpart[i];
  sm[threadIdx.x] = s;
  __syncthreads();
  for (int st = 128; st > 0; st >>= 1) {
    if (threadIdx.x < st) sm[threadIdx.x] += sm[threadIdx.x + st];
    __syncthreads();
  }
  if (threadIdx.x == 0)
    out[ZQ_ELEMS] = KLW * sm[0] / (float)(NBATCH * HW);
}

extern "C" void kernel_launch(void* const* d_in, const int* in_sizes, int n_in,
                              void* d_out, int out_size, void* d_ws, size_t ws_size,
                              hipStream_t stream) {
  const float* z  = (const float*)d_in[0];
  const float* pw = (const float*)d_in[1];
  const float* pb = (const float*)d_in[2];
  const float* ew = (const float*)d_in[3];
  const float* u  = (const float*)d_in[4];
  float* out    = (float*)d_out;
  float* klpart = (float*)d_ws;   // 2048 floats

  hipLaunchKernelGGL(gumbelq_fused, dim3(MTILES / 8), dim3(256), 0, stream,
                     z, pw, pb, ew, u, out, klpart);
  hipLaunchKernelGGL(kl_reduce, dim3(1), dim3(256), 0, stream, klpart, out);
}
